// CausalSelfAttention_35373350650457
// MI455X (gfx1250) — matrix operand order repaired and checked by reading, not compiled, not run on hardware
//
#include <hip/hip_runtime.h>
#include <hip/hip_bf16.h>

typedef __attribute__((ext_vector_type(16))) __bf16 v16bf;
typedef __attribute__((ext_vector_type(8)))  float  v8f;

#define Bc 2
#define Tc 2048
#define Cc 768
#define Hc 12
#define Dc 64

union FragBF { v16bf v; unsigned u[8]; };

__device__ __forceinline__ unsigned ld32(const __bf16* p) {
  return *reinterpret_cast<const unsigned*>(p);
}

// A-matrix fragment (16x32 bf16, row-major source, K-contiguous):
// VGPR r in 0..3 : K = 8g + 2r + p ; VGPR 4..7 : K = 16 + 8g + 2(r-4) + p
__device__ __forceinline__ v16bf load_a_frag(const __bf16* rowBase, int k0, int g) {
  FragBF f;
#pragma unroll
  for (int r = 0; r < 4; ++r) f.u[r]     = ld32(rowBase + k0 + 8*g + 2*r);
#pragma unroll
  for (int r = 0; r < 4; ++r) f.u[4 + r] = ld32(rowBase + k0 + 16 + 8*g + 2*r);
  return f.v;
}

// B-matrix fragment (32x16 bf16) from K-contiguous [N][K] source:
// lane n = column, VGPR r in 0..3 : K = 8g + 2r + p ; VGPR 4..7 : K = 16 + 8g + 2(r-4) + p
__device__ __forceinline__ v16bf load_b_frag(const __bf16* colBase, int k0, int g) {
  FragBF f;
#pragma unroll
  for (int r = 0; r < 8; ++r) f.u[r] = ld32(colBase + k0 + 8*g + 2*(r & 3) + 16*(r >> 2));
  return f.v;
}

__device__ __forceinline__ v8f wmma_bf16(v16bf a, v16bf b, v8f c) {
  return __builtin_amdgcn_wmma_f32_16x16x32_bf16(false, a, false, b, (short)0, c, false, false);
}

// ---------------------------------------------------------------- converts
__global__ void cvt_bf16_kernel(const float* __restrict__ in, __bf16* __restrict__ out, int n) {
  int i = blockIdx.x * blockDim.x + threadIdx.x;
  if (i < n) out[i] = (__bf16)in[i];
}

// in[K][N] f32 -> out[N][K] bf16
__global__ void transpose_bf16_kernel(const float* __restrict__ in, __bf16* __restrict__ out,
                                      int K, int N) {
  int i = blockIdx.x * blockDim.x + threadIdx.x;
  if (i >= K * N) return;
  int k = i / N, n = i % N;
  out[(size_t)n * K + k] = (__bf16)in[i];
}

// ---------------------------------------------------------------- GEMM
// C[m,n] = sum_k A[m,k] * Bt[n,k] + bias[n]   (wave tile 32x64, 2x4 WMMA accs)
// EPI==0: write f32 out[M,N].  EPI==1: scatter into Q/K/Vt bf16 (qkv split).
template <int EPI>
__global__ __launch_bounds__(256)
void gemm_bf16_kernel(const __bf16* __restrict__ A, const __bf16* __restrict__ Bt,
                      const float* __restrict__ bias, float* __restrict__ outF,
                      __bf16* __restrict__ q_out, __bf16* __restrict__ k_out,
                      __bf16* __restrict__ vt_out, int M, int N, int K) {
  const int lane = threadIdx.x & 31, lm = lane & 15, g = lane >> 4;
  const int wid  = blockIdx.x * (blockDim.x >> 5) + (threadIdx.x >> 5);
  const int ntn  = N >> 6;
  const int tm = wid / ntn, tn = wid % ntn;        // wave-uniform
  if (tm >= (M >> 5)) return;
  const int m0 = tm << 5, n0 = tn << 6;

  v8f acc[2][4] = {};
  const __bf16* aRow[2];
  const __bf16* bCol[4];
#pragma unroll
  for (int i = 0; i < 2; ++i) aRow[i] = A  + (size_t)(m0 + i * 16 + lm) * K;
#pragma unroll
  for (int j = 0; j < 4; ++j) bCol[j] = Bt + (size_t)(n0 + j * 16 + lm) * K;

  for (int k0 = 0; k0 < K; k0 += 32) {
    // prefetch one K-step ahead (speculative; dropped if OOB)
    __builtin_prefetch(aRow[0] + k0 + 32, 0, 0);
    __builtin_prefetch(bCol[0] + k0 + 32, 0, 0);
    __builtin_prefetch(bCol[2] + k0 + 32, 0, 0);

    v16bf a0 = load_a_frag(aRow[0], k0, g);
    v16bf a1 = load_a_frag(aRow[1], k0, g);
#pragma unroll
    for (int j = 0; j < 4; ++j) {
      v16bf bf = load_b_frag(bCol[j], k0, g);
      acc[0][j] = wmma_bf16(a0, bf, acc[0][j]);
      acc[1][j] = wmma_bf16(a1, bf, acc[1][j]);
    }
  }

#pragma unroll
  for (int i = 0; i < 2; ++i)
#pragma unroll
    for (int j = 0; j < 4; ++j)
#pragma unroll
      for (int v = 0; v < 8; ++v) {
        int m = m0 + i * 16 + v + 8 * g;       // C layout: row = v + 8g
        int n = n0 + j * 16 + lm;              // col = lane%16
        float val = acc[i][j][v] + bias[n];
        if (EPI == 0) {
          outF[(size_t)m * N + n] = val;
        } else {
          int b = m >> 11, t = m & (Tc - 1);
          int part = n / Cc, c = n % Cc;
          int h = c >> 6, d = c & 63;
          __bf16 bv = (__bf16)val;
          size_t bh = (size_t)b * Hc + h;
          if (part == 0)      q_out[(bh * Tc + t) * Dc + d]  = bv;
          else if (part == 1) k_out[(bh * Tc + t) * Dc + d]  = bv;
          else                vt_out[(bh * Dc + d) * Tc + t] = bv;   // V transposed
        }
      }
}

// ---------------------------------------------------------------- flash attention
// One wave = 16 query rows of one (b,h). Online softmax, causal.
// V tiles (64 d-rows x 32 kv, bf16) are double-buffered into wave-private LDS
// via the gfx1250 async global->LDS engine.
__device__ __forceinline__ void issue_v_copy(const __bf16* Vbase, int kv0,
                                             __bf16* dst, int lane) {
  // prior ds reads of the destination buffer must be complete: async LDS
  // writes are unordered vs the DS pipe.
  asm volatile("s_wait_dscnt 0x0" ::: "memory");
#pragma unroll
  for (int c2 = 0; c2 < 8; ++c2) {
    int chunk = c2 * 32 + lane;                 // 256 x 16B chunks = 4 KB tile
    int row   = chunk >> 2;                     // d row (0..63)
    int eoff  = (chunk & 3) * 8;                // kv element offset within row
    const __bf16* src = Vbase + (size_t)row * Tc + kv0 + eoff;
    unsigned ldsOff = (unsigned)(uintptr_t)(dst + chunk * 8);
    asm volatile("global_load_async_to_lds_b128 %0, %1, off"
                 :: "v"(ldsOff), "v"((unsigned long long)(uintptr_t)src)
                 : "memory");
  }
}

__global__ __launch_bounds__(256)
void attn_kernel(const __bf16* __restrict__ Qb, const __bf16* __restrict__ Kb,
                 const __bf16* __restrict__ Vt, __bf16* __restrict__ Yb) {
  __shared__ __bf16 ldsP[8][16 * 32];          // per-wave P staging (8 KB)
  __shared__ __bf16 ldsV[8][2][64 * 32];       // per-wave V double buffer (64 KB)
  const int lane = threadIdx.x & 31, lm = lane & 15, g = lane >> 4;
  const int widx = threadIdx.x >> 5;
  const int gw = blockIdx.x * 8 + widx;        // wave-uniform
  const int qt = gw & 127;                     // T/16 = 128 query tiles
  const int bh = gw >> 7;
  if (bh >= Bc * Hc) return;
  const int b = bh / Hc, h = bh % Hc;
  const int q0 = qt << 4;

  const __bf16* Qbase = Qb + (size_t)bh * Tc * Dc;
  const __bf16* Kbase = Kb + (size_t)bh * Tc * Dc;
  const __bf16* Vbase = Vt + (size_t)bh * Dc * Tc;

  v16bf qf[2];
  {
    const __bf16* qrow = Qbase + (size_t)(q0 + lm) * Dc;
    qf[0] = load_a_frag(qrow, 0, g);
    qf[1] = load_a_frag(qrow, 32, g);
  }

  v8f o[4] = {};
  float mrow[8], lrow[8];
#pragma unroll
  for (int v = 0; v < 8; ++v) { mrow[v] = -__builtin_inff(); lrow[v] = 0.f; }

  const int nkb = ((q0 + 15) >> 5) + 1;        // causal: only needed key blocks
  issue_v_copy(Vbase, 0, &ldsV[widx][0][0], lane);

  for (int kb = 0; kb < nkb; ++kb) {
    const int kv0 = kb << 5;
    const int buf = kb & 1;
    const bool havenext = (kb + 1) < nkb;
    if (havenext)
      issue_v_copy(Vbase, kv0 + 32, &ldsV[widx][buf ^ 1][0], lane);

    // S = Q @ K^T  (16 x 32 scores = 2 n-tiles, K-dim = 64 = 2 k-steps)
    v8f sc[2] = {};
#pragma unroll
    for (int nt = 0; nt < 2; ++nt) {
      const __bf16* kcol = Kbase + (size_t)(kv0 + nt * 16 + lm) * Dc;
      v16bf k0f = load_b_frag(kcol, 0, g);
      v16bf k1f = load_b_frag(kcol, 32, g);
      sc[nt] = wmma_bf16(qf[0], k0f, sc[nt]);
      sc[nt] = wmma_bf16(qf[1], k1f, sc[nt]);
    }

    // scale + causal mask + online softmax
    float p[2][8];
#pragma unroll
    for (int v = 0; v < 8; ++v) {
      const int mq = q0 + v + 8 * g;
      p[0][v] = (kv0 + lm      <= mq) ? sc[0][v] * 0.125f : -__builtin_inff();
      p[1][v] = (kv0 + 16 + lm <= mq) ? sc[1][v] * 0.125f : -__builtin_inff();
    }
#pragma unroll
    for (int v = 0; v < 8; ++v) {
      float rm = fmaxf(p[0][v], p[1][v]);
#pragma unroll
      for (int mk = 1; mk < 16; mk <<= 1) rm = fmaxf(rm, __shfl_xor(rm, mk, 32));
      float mnew = fmaxf(mrow[v], rm);
      float c = __expf(mrow[v] - mnew);
      mrow[v] = mnew;
      float p0 = __expf(p[0][v] - mnew);
      float p1 = __expf(p[1][v] - mnew);
      p[0][v] = p0; p[1][v] = p1;
      float rs = p0 + p1;
#pragma unroll
      for (int mk = 1; mk < 16; mk <<= 1) rs += __shfl_xor(rs, mk, 32);
      lrow[v] = lrow[v] * c + rs;
#pragma unroll
      for (int dt = 0; dt < 4; ++dt) o[dt][v] = o[dt][v] * c;
    }

    // stage P (C layout -> row-major LDS) then re-read in A layout
#pragma unroll
    for (int nt = 0; nt < 2; ++nt)
#pragma unroll
      for (int v = 0; v < 8; ++v)
        ldsP[widx][(v + 8 * g) * 32 + nt * 16 + lm] = (__bf16)p[nt][v];
    asm volatile("s_wait_dscnt 0x0" ::: "memory");

    v16bf pf;
    {
      const __bf16* prow = &ldsP[widx][lm * 32];
      FragBF f;
#pragma unroll
      for (int r = 0; r < 4; ++r) f.u[r]     = *reinterpret_cast<const unsigned*>(prow + 8*g + 2*r);
#pragma unroll
      for (int r = 0; r < 4; ++r) f.u[4 + r] = *reinterpret_cast<const unsigned*>(prow + 16 + 8*g + 2*r);
      pf = f.v;
    }

    // wait for this block's async V tile; async loads complete in-order, so
    // <=8 is safe while the next block's 8 copies are still in flight.
    if (havenext) asm volatile("s_wait_asynccnt 0x8" ::: "memory");
    else          asm volatile("s_wait_asynccnt 0x0" ::: "memory");

    // O += P @ V  (V fragments from LDS tile, 64B rows -> conflict-free)
#pragma unroll
    for (int dt = 0; dt < 4; ++dt) {
      const __bf16* vrow = &ldsV[widx][buf][(dt * 16 + lm) * 32];
      FragBF f;
#pragma unroll
      for (int r = 0; r < 8; ++r)
        f.u[r] = *reinterpret_cast<const unsigned*>(vrow + 8 * g + 2 * (r & 3) + 16 * (r >> 2));
      o[dt] = wmma_bf16(pf, f.v, o[dt]);
    }
  }

  // normalize and emit y as bf16 [B,T,C] (input to output projection)
  const size_t rowBase = ((size_t)b * Tc + q0) * Cc;
#pragma unroll
  for (int dt = 0; dt < 4; ++dt)
#pragma unroll
    for (int v = 0; v < 8; ++v) {
      float val = o[dt][v] / lrow[v];
      int t = v + 8 * g;
      Yb[rowBase + (size_t)t * Cc + h * 64 + dt * 16 + lm] = (__bf16)val;
    }
}

// ---------------------------------------------------------------- launch
extern "C" void kernel_launch(void* const* d_in, const int* in_sizes, int n_in,
                              void* d_out, int out_size, void* d_ws, size_t ws_size,
                              hipStream_t stream) {
  (void)in_sizes; (void)n_in; (void)out_size; (void)ws_size;
  const float* x      = (const float*)d_in[0];
  const float* w_attn = (const float*)d_in[1];
  const float* b_attn = (const float*)d_in[2];
  const float* w_proj = (const float*)d_in[3];
  const float* b_proj = (const float*)d_in[4];
  float* out = (float*)d_out;

  const int M  = Bc * Tc;   // 4096
  const int K  = Cc;        // 768
  const int N1 = 3 * Cc;    // 2304
  const int N2 = Cc;        // 768

  char* w = (char*)d_ws;
  auto carve = [&](size_t bytes) { char* p = w; w += (bytes + 255) & ~(size_t)255; return p; };
  __bf16* xb  = (__bf16*)carve((size_t)M * K * 2);
  __bf16* waT = (__bf16*)carve((size_t)N1 * K * 2);
  __bf16* wpT = (__bf16*)carve((size_t)N2 * K * 2);
  __bf16* Qb  = (__bf16*)carve((size_t)Bc * Hc * Tc * Dc * 2);
  __bf16* Kb  = (__bf16*)carve((size_t)Bc * Hc * Tc * Dc * 2);
  __bf16* Vt  = (__bf16*)carve((size_t)Bc * Hc * Dc * Tc * 2);
  __bf16* Yb  = (__bf16*)carve((size_t)M * Cc * 2);

  int nx = M * K;
  cvt_bf16_kernel<<<(nx + 255) / 256, 256, 0, stream>>>(x, xb, nx);
  transpose_bf16_kernel<<<(K * N1 + 255) / 256, 256, 0, stream>>>(w_attn, waT, K, N1);
  transpose_bf16_kernel<<<(K * N2 + 255) / 256, 256, 0, stream>>>(w_proj, wpT, K, N2);

  int tiles1 = (M / 32) * (N1 / 64);   // 4608 wave tiles
  gemm_bf16_kernel<1><<<(tiles1 + 7) / 8, 256, 0, stream>>>(
      xb, waT, b_attn, nullptr, Qb, Kb, Vt, M, N1, K);

  int nwaves = Bc * Hc * (Tc / 16);    // 3072 waves
  attn_kernel<<<nwaves / 8, 256, 0, stream>>>(Qb, Kb, Vt, Yb);

  int tiles2 = (M / 32) * (N2 / 64);   // 1536 wave tiles
  gemm_bf16_kernel<0><<<(tiles2 + 7) / 8, 256, 0, stream>>>(
      Yb, wpT, b_proj, out, nullptr, nullptr, nullptr, M, N2, K);
}
